// GAT_51616916963750
// MI455X (gfx1250) — matrix operand order repaired and checked by reading, not compiled, not run on hardware
//
#include <hip/hip_runtime.h>
#include <hip/hip_fp16.h>

#define N_NODES   100000
#define N_EDGES   1600000
#define F_DIM     128          // feature width of layers 0/1 (8 heads x 16)
#define HEADS     8
#define PER_HEAD  16
#define N_CLS     40
#define NEG_SLOPE 0.2f
#define EPS_SOFT  1e-16f

typedef __attribute__((ext_vector_type(16))) _Float16 v16h;
typedef __attribute__((ext_vector_type(8)))  _Float16 v8h;
typedef __attribute__((ext_vector_type(8)))  float    v8f;

// ---- monotonic uint encoding of float for atomic segment-max ----
static __device__ __forceinline__ unsigned f2ord(float f) {
    unsigned u = __float_as_uint(f);
    return (u & 0x80000000u) ? ~u : (u | 0x80000000u);
}
static __device__ __forceinline__ float ord2f(unsigned o) {
    unsigned u = (o & 0x80000000u) ? (o & 0x7FFFFFFFu) : ~o;
    return __uint_as_float(u);
}

// ---------------------------------------------------------------------------
// Pack W [128][nOutCols] (f32, row-major) into per-lane WMMA B fragments:
// frag[((kt*nTilesN + ntile)*32 + lane)*16 + i] = (f16) W[kt*32 + (lane>>4)*16 + i][ntile*16 + (lane&15)]
// ---------------------------------------------------------------------------
__global__ void prep_bfrag(const float* __restrict__ W, _Float16* __restrict__ frag,
                           int nTilesN, int nOutCols) {
    int idx = blockIdx.x * blockDim.x + threadIdx.x;
    int total = 4 * nTilesN * 32;
    if (idx >= total) return;
    int lane  = idx & 31;
    int t     = idx >> 5;
    int ntile = t % nTilesN;
    int kt    = t / nTilesN;
    int col   = ntile * 16 + (lane & 15);
    int kbase = kt * 32 + (lane >> 4) * 16;
    _Float16* dst = frag + (size_t)idx * 16;
    for (int i = 0; i < 16; ++i) {
        float v = (col < nOutCols) ? W[(size_t)(kbase + i) * nOutCols + col] : 0.0f;
        dst[i] = (_Float16)v;
    }
}

__global__ void cvt_to_f16(const float* __restrict__ src, _Float16* __restrict__ dst, int n) {
    int i = blockIdx.x * blockDim.x + threadIdx.x;
    if (i < n) dst[i] = (_Float16)src[i];
}

__global__ void zero_f32(float* __restrict__ p, int n) {
    int i = blockIdx.x * blockDim.x + threadIdx.x;
    if (i < n) p[i] = 0.0f;
}

// ---------------------------------------------------------------------------
// WMMA GEMM: Hout[N][ldH] = Ah[N][128](f16) x W(f16 frags) + bias, f32 out.
// One wave = one 16x16 tile; wave w of a block handles column tile w.
// gridDim.x = N/16 row tiles; blockDim.x = 32*nTilesN.
// ---------------------------------------------------------------------------
__global__ void gemm_wmma_bias(const _Float16* __restrict__ Ah,
                               const _Float16* __restrict__ Bfrag,
                               const float* __restrict__ bias,
                               float* __restrict__ Hout,
                               int nTilesN, int ldH, int nOutCols) {
    int lane    = threadIdx.x & 31;
    int ntile   = threadIdx.x >> 5;
    int rowTile = blockIdx.x;
    int row     = rowTile * 16 + (lane & 15);
    int kHalf   = lane >> 4;               // 0 or 1

    const _Float16* aBase = Ah + (size_t)row * F_DIM + kHalf * 16;
    const _Float16* bBase = Bfrag + ((size_t)ntile * 32 + lane) * 16;

    v8f c = {};
    #pragma unroll
    for (int kt = 0; kt < 4; ++kt) {
        v8h alo = *(const v8h*)(aBase + kt * 32);
        v8h ahi = *(const v8h*)(aBase + kt * 32 + 8);
        v16h a = __builtin_shufflevector(alo, ahi, 0,1,2,3,4,5,6,7,8,9,10,11,12,13,14,15);
        v16h b = *(const v16h*)(bBase + (size_t)kt * nTilesN * 32 * 16);
        c = __builtin_amdgcn_wmma_f32_16x16x32_f16(false, a, false, b,
                                                   (short)0, c, false, false);
    }

    int col   = ntile * 16 + (lane & 15);
    int mBase = rowTile * 16 + (lane >> 4) * 8;
    if (col < nOutCols) {
        float bv = bias[col];
        #pragma unroll
        for (int r = 0; r < 8; ++r)
            Hout[(size_t)(mBase + r) * ldH + col] = c[r] + bv;
    }
}

// ---------------------------------------------------------------------------
// Per-node attention terms + init of segment max/sum.
// ---------------------------------------------------------------------------
__global__ void att_init(const float* __restrict__ Hfeat,
                         const float* __restrict__ al, const float* __restrict__ ar,
                         float* __restrict__ a_l, float* __restrict__ a_r,
                         unsigned* __restrict__ m_ord, float* __restrict__ s,
                         int H, int C, int F) {
    int idx = blockIdx.x * blockDim.x + threadIdx.x;
    if (idx >= N_NODES * H) return;
    int n = idx / H, hd = idx % H;
    const float* hp = Hfeat + (size_t)n * F + hd * C;
    float suml = 0.f, sumr = 0.f;
    for (int c = 0; c < C; ++c) {
        float v = hp[c];
        suml += v * al[hd * C + c];
        sumr += v * ar[hd * C + c];
    }
    a_l[idx] = suml;
    a_r[idx] = sumr;
    m_ord[idx] = 0u;      // below f2ord(-inf): acts as -infinity
    s[idx] = 0.f;
}

// e = leaky_relu(a_l[src] + a_r[dst]); segment max via ordered-uint atomicMax
__global__ void edge_max(const int* __restrict__ ei,
                         const float* __restrict__ a_l, const float* __restrict__ a_r,
                         float* __restrict__ e_buf, unsigned* __restrict__ m_ord, int H) {
    int idx = blockIdx.x * blockDim.x + threadIdx.x;
    if (idx >= N_EDGES * H) return;
    int e = idx / H, hd = idx % H;
    int src = ei[e], dst = ei[N_EDGES + e];
    float v = a_l[(size_t)src * H + hd] + a_r[(size_t)dst * H + hd];
    v = v > 0.f ? v : NEG_SLOPE * v;
    e_buf[idx] = v;
    atomicMax(&m_ord[(size_t)dst * H + hd], f2ord(v));
}

// p = exp(e - m[dst]); segment sum
__global__ void edge_expsum(const int* __restrict__ ei,
                            float* __restrict__ e_buf, const unsigned* __restrict__ m_ord,
                            float* __restrict__ s, int H) {
    int idx = blockIdx.x * blockDim.x + threadIdx.x;
    if (idx >= N_EDGES * H) return;
    int e = idx / H, hd = idx % H;
    int dst = ei[N_EDGES + e];
    float m = ord2f(m_ord[(size_t)dst * H + hd]);
    float p = __expf(e_buf[idx] - m);
    e_buf[idx] = p;
    atomicAdd(&s[(size_t)dst * H + hd], p);
}

// acc[dst] += h[src] * (p / (s[dst]+eps)) per head
__global__ void edge_agg(const int* __restrict__ ei,
                         const float* __restrict__ e_buf, const float* __restrict__ s,
                         const float* __restrict__ Hfeat, float* __restrict__ acc,
                         int H, int C, int F) {
    int idx = blockIdx.x * blockDim.x + threadIdx.x;
    if (idx >= N_EDGES * H) return;
    int e = idx / H, hd = idx % H;
    int src = ei[e], dst = ei[N_EDGES + e];
    float alpha = e_buf[idx] / (s[(size_t)dst * H + hd] + EPS_SOFT);
    const float* hs = Hfeat + (size_t)src * F + hd * C;
    float*       op = acc   + (size_t)dst * F + hd * C;
    for (int c = 0; c < C; ++c)
        atomicAdd(&op[c], hs[c] * alpha);
}

// layers 0/1 epilogue: elu(acc + bias) -> f16 features for next GEMM
__global__ void finalize_elu_f16(const float* __restrict__ acc, const float* __restrict__ bias,
                                 _Float16* __restrict__ outh, int F) {
    int idx = blockIdx.x * blockDim.x + threadIdx.x;
    if (idx >= N_NODES * F) return;
    int c = idx % F;
    float v = acc[idx] + bias[c];
    v = v > 0.f ? v : expm1f(v);
    outh[idx] = (_Float16)v;
}

// final epilogue: log_softmax(acc + b2) over 40 classes (1 head, mean == identity)
__global__ void finalize_logsoftmax(const float* __restrict__ acc, const float* __restrict__ b2,
                                    float* __restrict__ out) {
    int n = blockIdx.x * blockDim.x + threadIdx.x;
    if (n >= N_NODES) return;
    float v[N_CLS];
    float m = -3.4e38f;
    for (int c = 0; c < N_CLS; ++c) {
        v[c] = acc[(size_t)n * N_CLS + c] + b2[c];
        m = v[c] > m ? v[c] : m;
    }
    float sum = 0.f;
    for (int c = 0; c < N_CLS; ++c) sum += __expf(v[c] - m);
    float ls = logf(sum);
    for (int c = 0; c < N_CLS; ++c)
        out[(size_t)n * N_CLS + c] = v[c] - m - ls;
}

// ---------------------------------------------------------------------------
extern "C" void kernel_launch(void* const* d_in, const int* in_sizes, int n_in,
                              void* d_out, int out_size, void* d_ws, size_t ws_size,
                              hipStream_t stream) {
    const float* x   = (const float*)d_in[0];
    const int*   ei  = (const int*)d_in[1];
    const float* W0  = (const float*)d_in[2];
    const float* lb0 = (const float*)d_in[3];
    const float* al0 = (const float*)d_in[4];
    const float* ar0 = (const float*)d_in[5];
    const float* b0  = (const float*)d_in[6];
    const float* W1  = (const float*)d_in[7];
    const float* lb1 = (const float*)d_in[8];
    const float* al1 = (const float*)d_in[9];
    const float* ar1 = (const float*)d_in[10];
    const float* b1  = (const float*)d_in[11];
    const float* W2  = (const float*)d_in[12];
    const float* lb2 = (const float*)d_in[13];
    const float* al2 = (const float*)d_in[14];
    const float* ar2 = (const float*)d_in[15];
    const float* b2  = (const float*)d_in[16];
    float* out = (float*)d_out;

    // ---- carve workspace ----
    size_t off = 0;
    auto carve = [&](size_t bytes) -> void* {
        void* p = (char*)d_ws + off;
        off += (bytes + 255) & ~(size_t)255;
        return p;
    };
    _Float16* frag0 = (_Float16*)carve((size_t)4 * 8 * 32 * 16 * 2);
    _Float16* frag1 = (_Float16*)carve((size_t)4 * 8 * 32 * 16 * 2);
    _Float16* frag2 = (_Float16*)carve((size_t)4 * 3 * 32 * 16 * 2);
    _Float16* xh    = (_Float16*)carve((size_t)N_NODES * F_DIM * 2);
    float*    hfeat = (float*)   carve((size_t)N_NODES * F_DIM * 4);
    float*    acc   = (float*)   carve((size_t)N_NODES * F_DIM * 4);
    float*    a_l   = (float*)   carve((size_t)N_NODES * HEADS * 4);
    float*    a_r   = (float*)   carve((size_t)N_NODES * HEADS * 4);
    unsigned* m_ord = (unsigned*)carve((size_t)N_NODES * HEADS * 4);
    float*    ssum  = (float*)   carve((size_t)N_NODES * HEADS * 4);
    float*    e_buf = (float*)   carve((size_t)N_EDGES * HEADS * 4);

    const int TPB = 256;
    const int rowTiles = N_NODES / 16;          // 6250 (exact)

    // ---- weight fragment packing (once per call) ----
    prep_bfrag<<<(4 * 8 * 32 + TPB - 1) / TPB, TPB, 0, stream>>>(W0, frag0, 8, F_DIM);
    prep_bfrag<<<(4 * 8 * 32 + TPB - 1) / TPB, TPB, 0, stream>>>(W1, frag1, 8, F_DIM);
    prep_bfrag<<<(4 * 3 * 32 + TPB - 1) / TPB, TPB, 0, stream>>>(W2, frag2, 3, N_CLS);

    cvt_to_f16<<<(N_NODES * F_DIM + TPB - 1) / TPB, TPB, 0, stream>>>(x, xh, N_NODES * F_DIM);

    // ---- layers 0 and 1 (identical shapes) ----
    const _Float16* frags[2] = {frag0, frag1};
    const float* lbs[2] = {lb0, lb1};
    const float* als[2] = {al0, al1};
    const float* ars[2] = {ar0, ar1};
    const float* bs[2]  = {b0, b1};

    for (int L = 0; L < 2; ++L) {
        gemm_wmma_bias<<<rowTiles, 8 * 32, 0, stream>>>(xh, frags[L], lbs[L],
                                                        hfeat, 8, F_DIM, F_DIM);
        att_init<<<(N_NODES * HEADS + TPB - 1) / TPB, TPB, 0, stream>>>(
            hfeat, als[L], ars[L], a_l, a_r, m_ord, ssum, HEADS, PER_HEAD, F_DIM);
        zero_f32<<<(N_NODES * F_DIM + TPB - 1) / TPB, TPB, 0, stream>>>(acc, N_NODES * F_DIM);
        edge_max<<<(N_EDGES * HEADS + TPB - 1) / TPB, TPB, 0, stream>>>(
            ei, a_l, a_r, e_buf, m_ord, HEADS);
        edge_expsum<<<(N_EDGES * HEADS + TPB - 1) / TPB, TPB, 0, stream>>>(
            ei, e_buf, m_ord, ssum, HEADS);
        edge_agg<<<(N_EDGES * HEADS + TPB - 1) / TPB, TPB, 0, stream>>>(
            ei, e_buf, ssum, hfeat, acc, HEADS, PER_HEAD, F_DIM);
        finalize_elu_f16<<<(N_NODES * F_DIM + TPB - 1) / TPB, TPB, 0, stream>>>(
            acc, bs[L], xh, F_DIM);
    }

    // ---- layer 2: 128 -> 40, single head, concat=False (mean over 1 head = id) ----
    gemm_wmma_bias<<<rowTiles, 3 * 32, 0, stream>>>(xh, frag2, lb2, hfeat, 3, N_CLS, N_CLS);
    att_init<<<(N_NODES + TPB - 1) / TPB, TPB, 0, stream>>>(
        hfeat, al2, ar2, a_l, a_r, m_ord, ssum, 1, N_CLS, N_CLS);
    zero_f32<<<(N_NODES * N_CLS + TPB - 1) / TPB, TPB, 0, stream>>>(acc, N_NODES * N_CLS);
    edge_max<<<(N_EDGES + TPB - 1) / TPB, TPB, 0, stream>>>(ei, a_l, a_r, e_buf, m_ord, 1);
    edge_expsum<<<(N_EDGES + TPB - 1) / TPB, TPB, 0, stream>>>(ei, e_buf, m_ord, ssum, 1);
    edge_agg<<<(N_EDGES + TPB - 1) / TPB, TPB, 0, stream>>>(
        ei, e_buf, ssum, hfeat, acc, 1, N_CLS, N_CLS);
    finalize_logsoftmax<<<(N_NODES + TPB - 1) / TPB, TPB, 0, stream>>>(acc, b2, out);
}
